// SelfAttention_9208409883120
// MI455X (gfx1250) — matrix operand order repaired and checked by reading, compiled, not run on hardware
//
#include <hip/hip_runtime.h>
#include <hip/hip_bf16.h>
#include <math.h>

// Problem constants (match reference).
#define BB   4
#define TT   8192
#define FF   512
#define HH   8
#define DHH  64
#define MM   (BB*TT)   // 32768 rows

typedef __bf16 v16bf __attribute__((ext_vector_type(16)));
typedef __bf16 v8bf  __attribute__((ext_vector_type(8)));
typedef float  v8f   __attribute__((ext_vector_type(8)));

// ---------------------------------------------------------------------------
// gfx1250 async copy: global -> LDS, 16 bytes per lane, tracked by ASYNCcnt.
// ---------------------------------------------------------------------------
__device__ __forceinline__ void async_copy_b128(unsigned lds_off, const void* g) {
  asm volatile("global_load_async_to_lds_b128 %0, %1, off"
               :: "v"(lds_off), "v"((unsigned long long)(uintptr_t)g)
               : "memory");
}
__device__ __forceinline__ void wait_asynccnt0() {
  asm volatile("s_wait_asynccnt 0x0" ::: "memory");
}
// Generic LDS pointer -> raw LDS byte offset (low 32 bits of generic address).
__device__ __forceinline__ unsigned lds_off_of(const void* p) {
  return (unsigned)(uintptr_t)p;
}

// ---------------------------------------------------------------------------
// K0: fp32 -> (hi, lo) bf16 split.  x ~= (float)hi + (float)lo, err ~2^-16.
// ---------------------------------------------------------------------------
__global__ void split_bf16_kernel(const float* __restrict__ x,
                                  __bf16* __restrict__ hi,
                                  __bf16* __restrict__ lo,
                                  int n) {
  int i = blockIdx.x * blockDim.x + threadIdx.x;
  if (i >= n) return;
  float v = x[i];
  __bf16 h = (__bf16)v;
  hi[i] = h;
  lo[i] = (__bf16)(v - (float)h);
}

// ---------------------------------------------------------------------------
// K1: pack a [FF x FF] fp32 weight into WMMA-B fragment-linear bf16 planes.
// Layout: idx = ((((kt*2 + plane)*NT + nt)*32 + lane)*16 + e)
//   plane 0 = hi, plane 1 = lo;  kt = k/32; within-tile K mapping (contiguous):
//   lanes 0-15 carry N (=nt*16+l16) with K=0..15 (e), lanes 16-31 K=16..31.
// A K-step's data for one 32-col block tile = two contiguous 2KB chunks
// (one per plane) -> single async b128 per thread when staging to LDS.
// ---------------------------------------------------------------------------
__global__ void pack_w_kernel(const float* __restrict__ W,
                              __bf16* __restrict__ pk) {
  int i = blockIdx.x * blockDim.x + threadIdx.x;   // over FF*FF
  if (i >= FF * FF) return;
  int n = i & (FF - 1);
  int k = i >> 9;                                  // FF == 512
  float v = W[i];
  __bf16 h = (__bf16)v;
  float r = v - (float)h;
  int kt     = k >> 5;
  int within = k & 31;
  int half   = within >> 4;
  int e      = within & 15;
  int nt     = n >> 4;
  int l16    = n & 15;
  int lane   = (half << 4) | l16;
  const int NT = FF / 16;
  size_t idx0 = ((((size_t)kt * 2 + 0) * NT + nt) * 32 + lane) * 16 + e;
  size_t idx1 = ((((size_t)kt * 2 + 1) * NT + nt) * 32 + lane) * 16 + e;
  pk[idx0] = h;
  pk[idx1] = (__bf16)r;
}

// ---------------------------------------------------------------------------
// A-fragment loader.  16-bit A (16x32): lanes 0-15 = M rows, half=0
// holds K=0..15 in v[0..15]; half=1 holds K=16..31 (matches packed B).
// Row-major A => two contiguous 16-byte loads per fragment.
// ---------------------------------------------------------------------------
__device__ __forceinline__ v16bf load_a_frag(const __bf16* __restrict__ A,
                                             int row, int kb, int half) {
  const __bf16* p = A + (size_t)row * FF + kb + (half << 4);
  v8bf x0 = *(const v8bf*)p;          // K = kb + 16*half .. +7
  v8bf x1 = *(const v8bf*)(p + 8);    // K = kb + 16*half + 8 .. +7
  v16bf r;
#pragma unroll
  for (int e = 0; e < 8; ++e) { r[e] = x0[e]; r[8 + e] = x1[e]; }
  return r;
}

// ---------------------------------------------------------------------------
// K2: C[MM x FF] = (Ahi+Alo)[MM x FF] @ (Bhi+Blo)[FF x FF], f32 accumulate.
// bf16x3: hi*hi + lo*hi + hi*lo (lo*lo ~2^-16, dropped).
// Block = 256 threads = 8 waves.  Each wave owns a 64x32 f32 tile (8 v8f
// accumulators).  The 32-col B tile (both planes, 4KB per K-step) is shared
// by all 8 waves and double-buffered in LDS via async b128 copies; each
// K-step issues the next stage's copy before computing, so the DMA overlaps
// the 24 WMMAs.  Grid: (FF/32, MM/512).
// ---------------------------------------------------------------------------
__global__ __launch_bounds__(256)
void gemm_bf16x3_kernel(const __bf16* __restrict__ Ahi,
                        const __bf16* __restrict__ Alo,
                        const __bf16* __restrict__ Bpk,   // packed hi/lo planes
                        float* __restrict__ C) {
  __shared__ __align__(32) __bf16 lbuf[2][2][1024];  // [parity][plane][2KB]

  const int tid  = threadIdx.x;
  const int lane = tid & 31;
  const int wave = tid >> 5;
  const int l16  = lane & 15;
  const int half = lane >> 4;
  const int m0   = blockIdx.y * 512 + wave * 64;
  const int n0   = blockIdx.x * 32;
  const int nt0  = n0 >> 4;
  const int NT   = FF / 16;
  const int NSTEP = FF / 32;

  // Staging role: threads 0-127 copy plane 0 (hi), 128-255 plane 1 (lo).
  const int spl  = tid >> 7;
  const int soff = (tid & 127) * 8;   // elements (16 bytes)

  // Prologue: stage kt = 0 into parity 0.
  {
    const __bf16* g = Bpk + (((size_t)0 * 2 + spl) * NT + nt0) * 512 + soff;
    async_copy_b128(lds_off_of(&lbuf[0][spl][soff]), g);
  }

  v8f acc[4][2] = {};

  for (int kt = 0; kt < NSTEP; ++kt) {
    const int par = kt & 1;
    wait_asynccnt0();
    __syncthreads();   // lbuf[par] filled; lbuf[par^1] no longer being read

    if (kt + 1 < NSTEP) {
      const __bf16* g =
          Bpk + ((((size_t)kt + 1) * 2 + spl) * NT + nt0) * 512 + soff;
      async_copy_b128(lds_off_of(&lbuf[par ^ 1][spl][soff]), g);
    }

    // B fragments for this wave (shared tile, read from LDS).
    v16bf bh0 = *(const v16bf*)&lbuf[par][0][(0 * 32 + lane) * 16];
    v16bf bh1 = *(const v16bf*)&lbuf[par][0][(1 * 32 + lane) * 16];
    v16bf bl0 = *(const v16bf*)&lbuf[par][1][(0 * 32 + lane) * 16];
    v16bf bl1 = *(const v16bf*)&lbuf[par][1][(1 * 32 + lane) * 16];

    const int kb = kt * 32;
#pragma unroll
    for (int i = 0; i < 4; ++i) {
      v16bf ah = load_a_frag(Ahi, m0 + i * 16 + l16, kb, half);
      v16bf al = load_a_frag(Alo, m0 + i * 16 + l16, kb, half);
      acc[i][0] = __builtin_amdgcn_wmma_f32_16x16x32_bf16(false, ah, false, bh0, (short)0, acc[i][0], false, false);
      acc[i][1] = __builtin_amdgcn_wmma_f32_16x16x32_bf16(false, ah, false, bh1, (short)0, acc[i][1], false, false);
      acc[i][0] = __builtin_amdgcn_wmma_f32_16x16x32_bf16(false, al, false, bh0, (short)0, acc[i][0], false, false);
      acc[i][1] = __builtin_amdgcn_wmma_f32_16x16x32_bf16(false, al, false, bh1, (short)0, acc[i][1], false, false);
      acc[i][0] = __builtin_amdgcn_wmma_f32_16x16x32_bf16(false, ah, false, bl0, (short)0, acc[i][0], false, false);
      acc[i][1] = __builtin_amdgcn_wmma_f32_16x16x32_bf16(false, ah, false, bl1, (short)0, acc[i][1], false, false);
    }
    __syncthreads();   // done reading lbuf[par] before it is refilled in kt+2
  }

  // C/D 16x16 f32 layout: VGPR r -> row r (lanes 0-15) / row 8+r (lanes 16-31)
#pragma unroll
  for (int i = 0; i < 4; ++i)
#pragma unroll
    for (int j = 0; j < 2; ++j)
#pragma unroll
      for (int r = 0; r < 8; ++r)
        C[(size_t)(m0 + i * 16 + half * 8 + r) * FF + n0 + j * 16 + l16] = acc[i][j][r];
}

// ---------------------------------------------------------------------------
// K3: bind[b,h][j] = sum_t sum_i K[t,i] * V[t,(j-i) mod 64].
// One block per (b,h); LDS-staged 16-token tiles; deterministic reduction.
// ---------------------------------------------------------------------------
__global__ __launch_bounds__(256)
void bind_kernel(const float* __restrict__ Km, const float* __restrict__ Vm,
                 float* __restrict__ bindOut) {
  __shared__ float lk[16][64];
  __shared__ float lv[16][64];
  __shared__ float red[256];
  const int bh  = blockIdx.x;     // 0..31
  const int b   = bh >> 3;
  const int h   = bh & 7;
  const int tid = threadIdx.x;
  const int j   = tid & 63;
  const int sub = tid >> 6;       // 0..3
  float acc = 0.f;
  const size_t base = ((size_t)b * TT) * FF + h * 64;
  for (int t0 = 0; t0 < TT; t0 += 16) {
#pragma unroll
    for (int c = 0; c < 4; ++c) {
      int lin = c * 256 + tid;
      int tt = lin >> 6, col = lin & 63;
      lk[tt][col] = Km[base + (size_t)(t0 + tt) * FF + col];
      lv[tt][col] = Vm[base + (size_t)(t0 + tt) * FF + col];
    }
    __syncthreads();
    for (int tt = sub; tt < 16; tt += 4) {
#pragma unroll
      for (int i = 0; i < 64; ++i)
        acc += lk[tt][i] * lv[tt][(j - i + 64) & 63];
    }
    __syncthreads();
  }
  red[tid] = acc;
  __syncthreads();
  if (tid < 64)
    bindOut[bh * 64 + tid] = red[tid] + red[tid + 64] + red[tid + 128] + red[tid + 192];
}

// ---------------------------------------------------------------------------
// K4: vp[t,j] = sum_m Q[t,m] * bind[(m+j)&63];
//     scale[t] = dot(v,vp) / (|v||vp|).   4 tokens / block, 64 threads each.
// ---------------------------------------------------------------------------
__global__ __launch_bounds__(256)
void score_kernel(const float* __restrict__ Q, const float* __restrict__ Vm,
                  const float* __restrict__ bind, float* __restrict__ scale) {
  __shared__ float qs[4][64], vs[4][64], bs[64];
  __shared__ float r0[256], r1[256], r2[256];
  const int bh  = blockIdx.y;
  const int b   = bh >> 3, h = bh & 7;
  const int tid = threadIdx.x;
  const int s   = tid >> 6, j = tid & 63;
  const int t   = blockIdx.x * 4 + s;
  if (tid < 64) bs[tid] = bind[bh * 64 + tid];
  const size_t rowBase = ((size_t)(b * TT + t)) * FF + h * 64;
  qs[s][j] = Q[rowBase + j];
  vs[s][j] = Vm[rowBase + j];
  __syncthreads();
  float vp = 0.f;
#pragma unroll
  for (int m = 0; m < 64; ++m)
    vp += qs[s][m] * bs[(m + j) & 63];
  float vj = vs[s][j];
  r0[tid] = vj * vp;
  r1[tid] = vj * vj;
  r2[tid] = vp * vp;
  for (int off = 32; off >= 1; off >>= 1) {
    __syncthreads();
    if (j < off) {
      r0[tid] += r0[tid + off];
      r1[tid] += r1[tid + off];
      r2[tid] += r2[tid + off];
    }
  }
  __syncthreads();
  if (j == 0)
    scale[(size_t)bh * TT + t] = r0[tid] * rsqrtf(r1[tid] * r2[tid]);
}

// ---------------------------------------------------------------------------
// K5: softmax over T per (b,h), in place.  One block per (b,h).
// ---------------------------------------------------------------------------
__global__ __launch_bounds__(256)
void softmax_kernel(float* __restrict__ sw) {
  __shared__ float red[256];
  const int tid = threadIdx.x;
  float* p = sw + (size_t)blockIdx.x * TT;
  float mx = -1e30f;
  for (int t = tid; t < TT; t += 256) mx = fmaxf(mx, p[t]);
  red[tid] = mx; __syncthreads();
  for (int off = 128; off >= 1; off >>= 1) {
    if (tid < off) red[tid] = fmaxf(red[tid], red[tid + off]);
    __syncthreads();
  }
  mx = red[0];
  __syncthreads();
  float sum = 0.f;
  for (int t = tid; t < TT; t += 256) sum += expf(p[t] - mx);
  red[tid] = sum; __syncthreads();
  for (int off = 128; off >= 1; off >>= 1) {
    if (tid < off) red[tid] += red[tid + off];
    __syncthreads();
  }
  const float inv = 1.0f / red[0];
  for (int t = tid; t < TT; t += 256) p[t] = expf(p[t] - mx) * inv;
}

// ---------------------------------------------------------------------------
// K6: WV[b,t,f] = weight[b, f/64, t] * V[b,t,f], split to bf16 hi/lo.
// ---------------------------------------------------------------------------
__global__ void scale_v_split_kernel(const float* __restrict__ v,
                                     const float* __restrict__ w,   // [B*H, T]
                                     __bf16* __restrict__ hi,
                                     __bf16* __restrict__ lo) {
  int i = blockIdx.x * blockDim.x + threadIdx.x;   // over MM*FF
  if (i >= MM * FF) return;
  int f  = i & (FF - 1);
  int bt = i >> 9;            // / FF
  int b  = bt >> 13;          // / TT
  int t  = bt & (TT - 1);
  int h  = f >> 6;
  float ww  = w[((b << 3) + h) * TT + t];
  float val = ww * v[i];
  __bf16 hh = (__bf16)val;
  hi[i] = hh;
  lo[i] = (__bf16)(val - (float)hh);
}

// ---------------------------------------------------------------------------
extern "C" void kernel_launch(void* const* d_in, const int* in_sizes, int n_in,
                              void* d_out, int out_size, void* d_ws, size_t ws_size,
                              hipStream_t stream) {
  const float* X  = (const float*)d_in[0];
  const float* Wq = (const float*)d_in[1];
  const float* Wk = (const float*)d_in[2];
  const float* Wv = (const float*)d_in[3];
  const float* Wo = (const float*)d_in[4];

  char* ws = (char*)d_ws;
  const size_t sBF  = (size_t)MM * FF * sizeof(__bf16);       // 32 MB
  const size_t sWpk = (size_t)2 * FF * FF * sizeof(__bf16);   // 1 MB (hi+lo)
  const size_t sF32 = (size_t)MM * FF * sizeof(float);        // 64 MB

  __bf16* Xhi = (__bf16*)(ws);
  __bf16* Xlo = (__bf16*)(ws + sBF);
  __bf16* Wpk[4];   // packed q, k, v, o (hi/lo planes interleaved per K-tile)
  for (int i = 0; i < 4; ++i) Wpk[i] = (__bf16*)(ws + 2 * sBF + (size_t)i * sWpk);
  float* Qb     = (float*)(ws + 2 * sBF + 4 * sWpk);
  float* Kb     = (float*)((char*)Qb + sF32);
  float* Vb     = (float*)((char*)Kb + sF32);
  float* bindB  = (float*)((char*)Vb + sF32);
  float* scaleB = (float*)((char*)bindB + (size_t)32 * 64 * sizeof(float));
  // total ws use: ~262 MB.  WVhi/WVlo reuse Xhi/Xlo (X dead after projections).

  {
    int n = MM * FF;
    split_bf16_kernel<<<(n + 255) / 256, 256, 0, stream>>>(X, Xhi, Xlo, n);
  }
  {
    int n = FF * FF;
    dim3 g((n + 255) / 256);
    pack_w_kernel<<<g, 256, 0, stream>>>(Wq, Wpk[0]);
    pack_w_kernel<<<g, 256, 0, stream>>>(Wk, Wpk[1]);
    pack_w_kernel<<<g, 256, 0, stream>>>(Wv, Wpk[2]);
    pack_w_kernel<<<g, 256, 0, stream>>>(Wo, Wpk[3]);
  }

  dim3 gg(FF / 32, MM / 512);
  gemm_bf16x3_kernel<<<gg, 256, 0, stream>>>(Xhi, Xlo, Wpk[0], Qb);
  gemm_bf16x3_kernel<<<gg, 256, 0, stream>>>(Xhi, Xlo, Wpk[1], Kb);
  gemm_bf16x3_kernel<<<gg, 256, 0, stream>>>(Xhi, Xlo, Wpk[2], Vb);

  bind_kernel<<<32, 256, 0, stream>>>(Kb, Vb, bindB);
  score_kernel<<<dim3(TT / 4, BB * HH), 256, 0, stream>>>(Qb, Vb, bindB, scaleB);
  softmax_kernel<<<32, 256, 0, stream>>>(scaleB);

  {
    int n = MM * FF;
    scale_v_split_kernel<<<(n + 255) / 256, 256, 0, stream>>>(Vb, scaleB, Xhi, Xlo);
  }
  gemm_bf16x3_kernel<<<gg, 256, 0, stream>>>(Xhi, Xlo, Wpk[3], (float*)d_out);
}
